// E3_GNNlayer_42528766165475
// MI455X (gfx1250) — matrix operand order repaired and checked by reading, not compiled, not run on hardware
//
#include <hip/hip_runtime.h>
#include <math.h>

typedef __attribute__((ext_vector_type(16))) _Float16 v16h;
typedef __attribute__((ext_vector_type(8)))  _Float16 v8h;
typedef __attribute__((ext_vector_type(8)))  float    v8f;

#define B_    4
#define L_    256
#define DN    256
#define DP    128
#define E_    65536
#define M0_   32
#define M1_   8
#define DIRR  56
#define WNUM_ 1664
#define NNODES 1024
#define LN_EPS_ 1e-5f

#define PW0_  0.15811388300841897f   // sqrt(1/40)
#define PW1_  0.25f
#define S3_   0.57735026918962576f   // 1/sqrt(3)
#define K121  0.54772255750516611f   // sqrt(3/10)

// ---------------------------------------------------------------------------
// Convert weights to transposed f16: wT[n][k], contiguous K for WMMA B frags.
// ---------------------------------------------------------------------------
__global__ void prep_weights(const float* __restrict__ fc1_w,
                             const float* __restrict__ fc2_w,
                             _Float16* __restrict__ fc1_wT,
                             _Float16* __restrict__ fc2_wT) {
    int idx = blockIdx.x * 256 + threadIdx.x;
    if (idx < 128 * 128) {
        int k = idx >> 7, n = idx & 127;
        fc1_wT[n * 128 + k] = (_Float16)fc1_w[k * 128 + n];
    }
    int idx2 = idx - 128 * 128;
    if (idx2 >= 0 && idx2 < 128 * WNUM_) {
        int k = idx2 / WNUM_, n = idx2 % WNUM_;
        fc2_wT[n * 128 + k] = (_Float16)fc2_w[idx2];
    }
}

// ---------------------------------------------------------------------------
// feats = concat(node @ proj_l0_w + b, l1_feats)  : (1024, 56)
// ---------------------------------------------------------------------------
__global__ void feats_kernel(const float* __restrict__ node,
                             const float* __restrict__ l1,
                             const float* __restrict__ w,
                             const float* __restrict__ b,
                             float* __restrict__ feats) {
    int n = blockIdx.x, t = threadIdx.x;
    if (t < 32) {
        float acc = b[t];
        const float* nr = node + (long)n * DN;
        for (int k = 0; k < DN; ++k) acc += nr[k] * w[k * 32 + t];
        feats[n * DIRR + t] = acc;
    } else if (t < 56) {
        feats[n * DIRR + t] = l1[n * 24 + (t - 32)];
    }
}

// ---------------------------------------------------------------------------
// Fused edge kernel: gather -> LN -> fc1 (WMMA) -> relu -> fc2 (WMMA, w kept
// in 208KB LDS) -> equivariant tensor product -> atomic segment sum.
// One block = 32 edges (two 16-row WMMA M-tiles), 8 waves as 2x4 grid.
// ---------------------------------------------------------------------------
__global__ __launch_bounds__(256) void edge_kernel(
    const float* __restrict__ pair, const float* __restrict__ edge_sh,
    const float* __restrict__ ln_g, const float* __restrict__ ln_b,
    const float* __restrict__ fc1_b, const float* __restrict__ fc2_b,
    const _Float16* __restrict__ fc1_wT, const _Float16* __restrict__ fc2_wT,
    const int* __restrict__ edge_src, const int* __restrict__ edge_dst,
    const float* __restrict__ feats, float* __restrict__ sums,
    float* __restrict__ cnt)
{
    extern __shared__ char smem[];
    _Float16* normbuf = (_Float16*)smem;            // 32 x 128 f16 =   8 KB
    _Float16* hbuf    = (_Float16*)(smem + 8192);   // 32 x 128 f16 =   8 KB
    float*    wbuf    = (float*)(smem + 16384);     // 32 x 1664 f32 = 208 KB

    const int tid = threadIdx.x;
    const int e0  = blockIdx.x * 32;

    // ---- Phase 1: gather pair rows, LayerNorm, write f16 to LDS ----------
    {
        int le = tid >> 3, r = tid & 7;            // 8 threads per edge
        int e  = e0 + le;
        int src = edge_src[e], dst = edge_dst[e];
        long prow = (long)src * L_ + (dst & (L_ - 1));   // (b*L+i)*L + j
        const float* pf = pair + prow * DP + r * 16;
        float v[16];
        float s = 0.f, ss = 0.f;
        for (int c = 0; c < 16; ++c) { float x = pf[c]; v[c] = x; s += x; ss += x * x; }
        s += __shfl_xor(s, 1);  ss += __shfl_xor(ss, 1);
        s += __shfl_xor(s, 2);  ss += __shfl_xor(ss, 2);
        s += __shfl_xor(s, 4);  ss += __shfl_xor(ss, 4);
        float mu   = s * (1.f / DP);
        float var  = ss * (1.f / DP) - mu * mu;
        float rstd = rsqrtf(var + LN_EPS_);
        for (int c = 0; c < 16; ++c) {
            int col = r * 16 + c;
            float h = (v[c] - mu) * rstd * ln_g[col] + ln_b[col];
            normbuf[le * DP + col] = (_Float16)h;
        }
    }
    __syncthreads();

    const int lane = tid & 31, wid = tid >> 5;
    const int wm = wid & 1, wn = wid >> 1;      // wave grid 2 (M) x 4 (N)
    const int hl = lane & 15, hi = lane >> 4;   // WMMA half-lane / half-sel
    union AF { v16h v; v8h h[2]; };

    // ---- Phase 2: fc1 = relu(LN(x) @ W1 + b1) via WMMA f16 ----------------
    AF afr[4];
    {
        const v8h* base = (const v8h*)(normbuf + (wm * 16 + hl) * DP);
        for (int kb = 0; kb < 4; ++kb) {        // A 16x32 frag: K contiguous k0+16*hi .. +15
            afr[kb].h[0] = base[kb * 4 + hi * 2];
            afr[kb].h[1] = base[kb * 4 + hi * 2 + 1];
        }
    }
    for (int nt = wn * 2; nt < wn * 2 + 2; ++nt) {
        int n0 = nt * 16;
        float bias = fc1_b[n0 + hl];
        v8f c; for (int i = 0; i < 8; ++i) c[i] = bias;
        for (int kb = 0; kb < 4; ++kb) {
            v16h b = *(const v16h*)(fc1_wT + (n0 + hl) * DP + kb * 32 + hi * 16);
            c = __builtin_amdgcn_wmma_f32_16x16x32_f16(false, afr[kb].v, false, b,
                                                       (short)0, c, false, false);
        }
        for (int r = 0; r < 8; ++r) {
            float hv = c[r] > 0.f ? c[r] : 0.f;   // ReLU
            hbuf[(wm * 16 + hi * 8 + r) * DP + n0 + hl] = (_Float16)hv;
        }
    }
    __syncthreads();

    // ---- Phase 3: fc2 = h @ W2 + b2 via WMMA, 104 N-tiles into LDS --------
    {
        const v8h* base = (const v8h*)(hbuf + (wm * 16 + hl) * DP);
        for (int kb = 0; kb < 4; ++kb) {
            afr[kb].h[0] = base[kb * 4 + hi * 2];
            afr[kb].h[1] = base[kb * 4 + hi * 2 + 1];
        }
    }
    for (int nt = wn; nt < 104; nt += 4) {
        int n0 = nt * 16;
        if (nt + 4 < 104)                        // L2 prefetch next weight tile
            __builtin_prefetch(fc2_wT + (n0 + 64 + hl) * DP, 0, 1);
        float bias = fc2_b[n0 + hl];
        v8f c; for (int i = 0; i < 8; ++i) c[i] = bias;
        for (int kb = 0; kb < 4; ++kb) {
            v16h b = *(const v16h*)(fc2_wT + (n0 + hl) * DP + kb * 32 + hi * 16);
            c = __builtin_amdgcn_wmma_f32_16x16x32_f16(false, afr[kb].v, false, b,
                                                       (short)0, c, false, false);
        }
        for (int r = 0; r < 8; ++r)
            wbuf[(wm * 16 + hi * 8 + r) * WNUM_ + n0 + hl] = c[r];
    }
    __syncthreads();

    // ---- Phase 4: per-edge equivariant tensor product + segment sum -------
    {
        int le = tid >> 3, r = tid & 7;          // 8 threads per edge
        int e  = e0 + le;
        int src = edge_src[e], dst = edge_dst[e];
        const float* x  = feats + (long)dst * DIRR;
        const float* sh = edge_sh + (long)e * 9;
        const float* w  = wbuf + le * WNUM_;
        float sh0 = sh[0];
        float s1x = sh[1], s1y = sh[2], s1z = sh[3];
        float s2a = sh[4], s2b = sh[5], s2c = sh[6], s2d = sh[7], s2e = sh[8];
        // Cm[a][m] = sum_b w3j[a][b][m] * sh2[b]  (sparse 3j hardcoded)
        float CmXX = K121 * (-(1.f / 3.f) * s2c - S3_ * s2e);
        float CmXY = K121 * (S3_ * s2b);
        float CmXZ = K121 * (S3_ * s2a);
        float CmYX = CmXY;
        float CmYY = K121 * ((2.f / 3.f) * s2c);
        float CmYZ = K121 * (S3_ * s2d);
        float CmZX = CmXZ;
        float CmZY = CmYZ;
        float CmZZ = K121 * (-(1.f / 3.f) * s2c + S3_ * s2e);
        float* srow = sums + (long)src * DIRR;

        // out0 : this thread owns columns r*4 .. r*4+3
        for (int j = 0; j < 4; ++j) {
            int wo = r * 4 + j;
            float acc = 0.f;
            for (int u = 0; u < 32; ++u) acc += x[u] * sh0 * w[u * 32 + wo];
            float acc2 = 0.f;
            for (int u = 0; u < 8; ++u) {
                float d = x[32 + u * 3] * s1x + x[33 + u * 3] * s1y + x[34 + u * 3] * s1z;
                acc2 += d * w[1344 + u * 32 + wo];
            }
            atomicAdd(&srow[wo], PW0_ * (acc + S3_ * acc2));
        }
        // out1 : this thread owns output channel wo = r (3 components)
        {
            int wo = r;
            float a0 = 0.f;
            for (int u = 0; u < 32; ++u) a0 += x[u] * w[1024 + u * 8 + wo];
            float t3x = 0.f, t3y = 0.f, t3z = 0.f;
            float t5x = 0.f, t5y = 0.f, t5z = 0.f;
            for (int u = 0; u < 8; ++u) {
                float xx = x[32 + u * 3], xy = x[33 + u * 3], xz = x[34 + u * 3];
                float w3 = w[1280 + u * 8 + wo];
                t3x += xx * w3; t3y += xy * w3; t3z += xz * w3;
                float t5ux = xx * CmXX + xy * CmYX + xz * CmZX;
                float t5uy = xx * CmXY + xy * CmYY + xz * CmZY;
                float t5uz = xx * CmXZ + xy * CmYZ + xz * CmZZ;
                float w5 = w[1600 + u * 8 + wo];
                t5x += t5ux * w5; t5y += t5uy * w5; t5z += t5uz * w5;
            }
            atomicAdd(&srow[32 + wo * 3 + 0], PW1_ * (S3_ * s1x * a0 + S3_ * sh0 * t3x + t5x));
            atomicAdd(&srow[32 + wo * 3 + 1], PW1_ * (S3_ * s1y * a0 + S3_ * sh0 * t3y + t5y));
            atomicAdd(&srow[32 + wo * 3 + 2], PW1_ * (S3_ * s1z * a0 + S3_ * sh0 * t3z + t5z));
        }
        if (r == 0) atomicAdd(&cnt[src], 1.0f);
    }
}

// ---------------------------------------------------------------------------
// finalize: mean, residuals, node projection
// ---------------------------------------------------------------------------
__global__ void finalize_kernel(const float* __restrict__ sums,
                                const float* __restrict__ cnt,
                                const float* __restrict__ node,
                                const float* __restrict__ l1,
                                const float* __restrict__ pw,
                                const float* __restrict__ pb,
                                float* __restrict__ out)
{
    __shared__ float so[DIRR];
    int n = blockIdx.x, t = threadIdx.x;
    if (t < DIRR) {
        float c = cnt[n]; c = c > 1.f ? c : 1.f;
        so[t] = sums[n * DIRR + t] / c;
    }
    __syncthreads();
    float acc = pb[t];
    for (int u = 0; u < 32; ++u) acc += so[u] * pw[u * DN + t];
    out[(long)n * DN + t] = acc + node[(long)n * DN + t];
    if (t < 24)
        out[(long)NNODES * DN + n * 24 + t] = so[32 + t] + l1[n * 24 + t];
}

// ---------------------------------------------------------------------------
extern "C" void kernel_launch(void* const* d_in, const int* in_sizes, int n_in,
                              void* d_out, int out_size, void* d_ws, size_t ws_size,
                              hipStream_t stream) {
    (void)in_sizes; (void)n_in; (void)out_size; (void)ws_size;
    const float* node = (const float*)d_in[0];
    const float* pair = (const float*)d_in[1];
    const float* l1   = (const float*)d_in[2];
    const float* sh   = (const float*)d_in[3];
    const float* pl0w = (const float*)d_in[4];
    const float* pl0b = (const float*)d_in[5];
    const float* pnw  = (const float*)d_in[6];
    const float* pnb  = (const float*)d_in[7];
    const float* lng  = (const float*)d_in[8];
    const float* lnb  = (const float*)d_in[9];
    const float* fc1w = (const float*)d_in[10];
    const float* fc1b = (const float*)d_in[11];
    const float* fc2w = (const float*)d_in[12];
    const float* fc2b = (const float*)d_in[13];
    const int*   esrc = (const int*)d_in[15];
    const int*   edst = (const int*)d_in[16];

    char* ws = (char*)d_ws;
    float*    sums = (float*)ws;                  // 1024*56*4 = 229376 B
    float*    cntp = (float*)(ws + 229376);       // 1024*4    =   4096 B
    float*    fts  = (float*)(ws + 233472);       // 1024*56*4 = 229376 B
    _Float16* fc1T = (_Float16*)(ws + 462848);    // 128*128*2 =  32768 B
    _Float16* fc2T = (_Float16*)(ws + 495616);    // 1664*128*2= 425984 B

    hipMemsetAsync(sums, 0, 229376 + 4096, stream);
    prep_weights<<<896, 256, 0, stream>>>(fc1w, fc2w, fc1T, fc2T);
    feats_kernel<<<NNODES, 64, 0, stream>>>(node, l1, pl0w, pl0b, fts);

    size_t shmem = 16384 + (size_t)32 * WNUM_ * 4;   // 229376 B (CDNA5 320KB LDS)
    edge_kernel<<<E_ / 32, 256, shmem, stream>>>(pair, sh, lng, lnb, fc1b, fc2b,
                                                 fc1T, fc2T, esrc, edst, fts,
                                                 sums, cntp);
    finalize_kernel<<<NNODES, 256, 0, stream>>>(sums, cntp, node, l1, pnw, pnb,
                                                (float*)d_out);
}
